// USPAttention_3942779977791
// MI455X (gfx1250) — matrix operand order repaired and checked by reading, not compiled, not run on hardware
//
#include <hip/hip_runtime.h>

typedef _Float16 f16;
typedef __attribute__((ext_vector_type(16))) _Float16 v16h;
typedef __attribute__((ext_vector_type(8)))  _Float16 v8h;
typedef __attribute__((ext_vector_type(8)))  float    v8f;
typedef unsigned int u32x4 __attribute__((ext_vector_type(4)));
typedef int          i32x4 __attribute__((ext_vector_type(4)));
typedef int          i32x8 __attribute__((ext_vector_type(8)));

#define B_      2
#define S_      2048
#define H_      16
#define D_      64
#define BLOCK_M 128   // 8 waves x 16 rows
#define BLOCK_N 64    // keys per iteration
#define LPAD    72    // f16 LDS row stride (halfs): 144B rows, 16B aligned, conflict-free
#define FPAD    68    // f32 staging row stride (floats): TDM pads 4 DWORDs per 64

#if __has_builtin(__builtin_amdgcn_tensor_load_to_lds) && __has_builtin(__builtin_amdgcn_s_wait_tensorcnt)
#define USE_TDM 1
#else
#define USE_TDM 0
#endif

union V16 { v16h v; v8h h[2]; f16 e[16]; };

__device__ __forceinline__ float fast_exp2(float x) {
#if __has_builtin(__builtin_amdgcn_exp2f)
  return __builtin_amdgcn_exp2f(x);      // bare v_exp_f32 (TRANS, co-executes with WMMA)
#else
  return exp2f(x);
#endif
}

__device__ __forceinline__ float hredmax16(float x) {
#pragma unroll
  for (int m = 1; m < 16; m <<= 1) x = fmaxf(x, __shfl_xor(x, m, 32));
  return x;
}

#if USE_TDM
// Issue one TDM 2D tile load: 64 rows x 64 f32, row stride H_*D_ floats in memory,
// packed to LDS with 4-DWORD pad per 64-DWORD row (=> FPAD-float LDS rows).
__device__ __forceinline__ void tdm_load_tile_f32(unsigned lds_byte_addr, const float* gsrc) {
  const unsigned long long ga = (unsigned long long)(size_t)gsrc;
  const u32x4 g0 = {
    1u,                                                    // count=1, user descriptor
    lds_byte_addr,                                         // lds_addr
    (unsigned)(ga & 0xffffffffull),                        // global_addr[31:0]
    (unsigned)((ga >> 32) & 0x1ffffffull) | 0x80000000u    // global_addr[56:32] | type=2
  };
  const i32x8 g1 = {
    (2 << 16) | (1 << 20) | (5 << 22) | (3 << 25),         // data_size=4B, pad_en, every 64 DW, +4 DW
    (D_ << 16),                                            // tensor_dim0 = 64  (bits 79:48)
    (BLOCK_N << 16),                                       // tensor_dim1 = 64  (bits 111:80)
    (D_ << 16),                                            // tile_dim0 = 64    (bits 127:112)
    BLOCK_N,                                               // tile_dim1 = 64, tile_dim2 = 0
    (int)((size_t)H_ * D_),                                // tensor_dim0_stride = 1024
    0, 0                                                   // stride0 hi / dim1_stride (unused, 2D)
  };
  const i32x4 gz = {0, 0, 0, 0};
#if __clang_major__ >= 23
  const i32x8 gz8 = {0, 0, 0, 0, 0, 0, 0, 0};
  __builtin_amdgcn_tensor_load_to_lds(g0, g1, gz, gz, gz8, 0);
#else
  __builtin_amdgcn_tensor_load_to_lds(g0, g1, gz, gz, 0);
#endif
}
#endif

__global__ __launch_bounds__(256)
void fa_fwd_kernel(const float* __restrict__ Q, const float* __restrict__ K,
                   const float* __restrict__ V, float* __restrict__ O)
{
  __shared__ __align__(32) f16 Kh[BLOCK_N * LPAD];   // [n][d]  f16 K block
  __shared__ __align__(32) f16 Vh[D_ * LPAD];        // [d][k]  f16 V block (transposed)
  __shared__ __align__(32) f16 Ph[8 * 16 * LPAD];    // per-wave P tile [wave][m][col]
#if USE_TDM
  __shared__ __align__(16) float Kf32[BLOCK_N * FPAD];  // TDM-staged raw f32 K block
  __shared__ __align__(16) float Vf32[BLOCK_N * FPAD];  // TDM-staged raw f32 V block
#endif

  const int tid  = threadIdx.x;
  const int wave = tid >> 5;
  const int lane = tid & 31;
  const int hi   = lane >> 4;     // half-wave: 0 -> lanes 0-15, 1 -> lanes 16-31
  const int ln   = lane & 15;

  const int bh = blockIdx.y;
  const int b  = bh >> 4;         // H_ == 16
  const int h  = bh & 15;
  const int m0 = blockIdx.x * BLOCK_M + wave * 16;

  const size_t rowstride = (size_t)H_ * D_;
  const float qscale = 0.125f * 1.44269504088896340736f;   // 1/sqrt(D) * log2(e)

  // ---- Q tile -> WMMA A-layout registers (f16, pre-scaled) ---------------
  V16 qa[2];
  {
    const float* qp = Q + (((size_t)b * S_ + (m0 + ln)) * H_ + h) * D_;
#pragma unroll
    for (int c = 0; c < 2; ++c) {
#pragma unroll
      for (int g = 0; g < 2; ++g) {
        const float* sp = qp + c * 32 + hi * 16 + g * 8;
        float4 f0 = *(const float4*)(sp);
        float4 f1 = *(const float4*)(sp + 4);
        const int e0 = g * 8;
        qa[c].e[e0 + 0] = (f16)(f0.x * qscale);
        qa[c].e[e0 + 1] = (f16)(f0.y * qscale);
        qa[c].e[e0 + 2] = (f16)(f0.z * qscale);
        qa[c].e[e0 + 3] = (f16)(f0.w * qscale);
        qa[c].e[e0 + 4] = (f16)(f1.x * qscale);
        qa[c].e[e0 + 5] = (f16)(f1.y * qscale);
        qa[c].e[e0 + 6] = (f16)(f1.z * qscale);
        qa[c].e[e0 + 7] = (f16)(f1.w * qscale);
      }
    }
  }

  // ones B-matrix: acc[4] = P * 1 accumulates the softmax denominator via WMMA
  V16 vone;
#pragma unroll
  for (int i = 0; i < 16; ++i) vone.e[i] = (f16)1.0f;

  v8f acc[5] = {};                 // [0..3]: O d-tiles; [4]: running row-sum l_i
  float mi[8];
#pragma unroll
  for (int r = 0; r < 8; ++r) mi[r] = -1.0e30f;

  const float* Kbase = K + (((size_t)b * S_) * H_ + h) * D_;
  const float* Vbase = V + (((size_t)b * S_) * H_ + h) * D_;

#if USE_TDM
  const unsigned kf_lds = (unsigned)(size_t)(void*)Kf32;
  const unsigned vf_lds = (unsigned)(size_t)(void*)Vf32;
  const int wave_s = __builtin_amdgcn_readfirstlane(wave);  // scalar branch: TDM ignores EXEC
  if (wave_s == 0) {
    tdm_load_tile_f32(kf_lds, Kbase);
    tdm_load_tile_f32(vf_lds, Vbase);
  }
#endif

  for (int kb = 0; kb < S_; kb += BLOCK_N) {
#if USE_TDM
    if (wave_s == 0) __builtin_amdgcn_s_wait_tensorcnt(0);
    __syncthreads();   // TDM-staged f32 block visible; prev compute done (Kh/Vh free)
    {
      // convert f32 -> f16 tiles: Kh[n][d], Vh[d][k] (transposed)
      const int r  = tid >> 2;
      const int qd = (tid & 3) * 16;
      const float* kp = &Kf32[r * FPAD + qd];
      const float* vp = &Vf32[r * FPAD + qd];
#else
    __syncthreads();
    {
      const int r  = tid >> 2;
      const int qd = (tid & 3) * 16;
      const float* kp = Kbase + (size_t)(kb + r) * rowstride + qd;
      const float* vp = Vbase + (size_t)(kb + r) * rowstride + qd;
#endif
      float4 k0 = *(const float4*)(kp);
      float4 k1 = *(const float4*)(kp + 4);
      float4 k2 = *(const float4*)(kp + 8);
      float4 k3 = *(const float4*)(kp + 12);
      V16 kv;
      kv.e[0]=(f16)k0.x;  kv.e[1]=(f16)k0.y;  kv.e[2]=(f16)k0.z;  kv.e[3]=(f16)k0.w;
      kv.e[4]=(f16)k1.x;  kv.e[5]=(f16)k1.y;  kv.e[6]=(f16)k1.z;  kv.e[7]=(f16)k1.w;
      kv.e[8]=(f16)k2.x;  kv.e[9]=(f16)k2.y;  kv.e[10]=(f16)k2.z; kv.e[11]=(f16)k2.w;
      kv.e[12]=(f16)k3.x; kv.e[13]=(f16)k3.y; kv.e[14]=(f16)k3.z; kv.e[15]=(f16)k3.w;
      *(v8h*)&Kh[r * LPAD + qd]     = kv.h[0];
      *(v8h*)&Kh[r * LPAD + qd + 8] = kv.h[1];

      float4 w0 = *(const float4*)(vp);
      float4 w1 = *(const float4*)(vp + 4);
      float4 w2 = *(const float4*)(vp + 8);
      float4 w3 = *(const float4*)(vp + 12);
      V16 vv;
      vv.e[0]=(f16)w0.x;  vv.e[1]=(f16)w0.y;  vv.e[2]=(f16)w0.z;  vv.e[3]=(f16)w0.w;
      vv.e[4]=(f16)w1.x;  vv.e[5]=(f16)w1.y;  vv.e[6]=(f16)w1.z;  vv.e[7]=(f16)w1.w;
      vv.e[8]=(f16)w2.x;  vv.e[9]=(f16)w2.y;  vv.e[10]=(f16)w2.z; vv.e[11]=(f16)w2.w;
      vv.e[12]=(f16)w3.x; vv.e[13]=(f16)w3.y; vv.e[14]=(f16)w3.z; vv.e[15]=(f16)w3.w;
#pragma unroll
      for (int i = 0; i < 16; ++i) Vh[(qd + i) * LPAD + r] = vv.e[i];
    }
    __syncthreads();   // f16 tiles ready; f32 staging buffers free for next DMA

#if USE_TDM
    if (wave_s == 0 && kb + BLOCK_N < S_) {     // DMA next block, overlapped with compute
      tdm_load_tile_f32(kf_lds, Kbase + (size_t)(kb + BLOCK_N) * rowstride);
      tdm_load_tile_f32(vf_lds, Vbase + (size_t)(kb + BLOCK_N) * rowstride);
    }
#endif

    // ---- S = Q * K^T : 4 n-tiles x (2 chunks of K=32) --------------------
    v8f sc[4];
#pragma unroll
    for (int j = 0; j < 4; ++j) {
      const f16* kpv = &Kh[(j * 16 + ln) * LPAD + hi * 16];
      V16 bk0, bk1;
      bk0.h[0] = *(const v8h*)(kpv);
      bk0.h[1] = *(const v8h*)(kpv + 8);
      bk1.h[0] = *(const v8h*)(kpv + 32);
      bk1.h[1] = *(const v8h*)(kpv + 40);
      v8f z = {};
      z = __builtin_amdgcn_wmma_f32_16x16x32_f16(false, qa[0].v, false, bk0.v, (short)0, z, false, false);
      z = __builtin_amdgcn_wmma_f32_16x16x32_f16(false, qa[1].v, false, bk1.v, (short)0, z, false, false);
      sc[j] = z;
    }

    // ---- online softmax (base-2; scale folded into Q) --------------------
    float alpha[8];
#pragma unroll
    for (int r = 0; r < 8; ++r) {
      float x = fmaxf(fmaxf(sc[0][r], sc[1][r]), fmaxf(sc[2][r], sc[3][r]));
      x = hredmax16(x);
      float mn = fmaxf(mi[r], x);
      alpha[r] = fast_exp2(mi[r] - mn);
      mi[r] = mn;
    }
#pragma unroll
    for (int j = 0; j < 4; ++j)
#pragma unroll
      for (int r = 0; r < 8; ++r)
        sc[j][r] = fast_exp2(sc[j][r] - mi[r]);
#pragma unroll
    for (int r = 0; r < 8; ++r)
#pragma unroll
      for (int t = 0; t < 5; ++t) acc[t][r] *= alpha[r];

    // ---- P: C-layout -> LDS -> A-layout ----------------------------------
    f16* pw = &Ph[wave * 16 * LPAD];
#pragma unroll
    for (int j = 0; j < 4; ++j)
#pragma unroll
      for (int r = 0; r < 8; ++r)
        pw[(r + hi * 8) * LPAD + (j * 16 + ln)] = (f16)sc[j][r];

    asm volatile("s_wait_dscnt 0" ::: "memory");   // same-wave cross-lane LDS RAW

    V16 pa[2];
#pragma unroll
    for (int c = 0; c < 2; ++c) {
      const f16* pp = &pw[ln * LPAD + c * 32 + hi * 16];
      pa[c].h[0] = *(const v8h*)(pp);
      pa[c].h[1] = *(const v8h*)(pp + 8);
    }

    // ---- O += P*V (4 d-tiles) and l += P*1 (ones column trick) -----------
#pragma unroll
    for (int t = 0; t < 4; ++t) {
      const f16* vpv = &Vh[(t * 16 + ln) * LPAD + hi * 16];
      V16 vb0, vb1;
      vb0.h[0] = *(const v8h*)(vpv);
      vb0.h[1] = *(const v8h*)(vpv + 8);
      vb1.h[0] = *(const v8h*)(vpv + 32);
      vb1.h[1] = *(const v8h*)(vpv + 40);
      acc[t] = __builtin_amdgcn_wmma_f32_16x16x32_f16(false, pa[0].v, false, vb0.v, (short)0, acc[t], false, false);
      acc[t] = __builtin_amdgcn_wmma_f32_16x16x32_f16(false, pa[1].v, false, vb1.v, (short)0, acc[t], false, false);
    }
    acc[4] = __builtin_amdgcn_wmma_f32_16x16x32_f16(false, pa[0].v, false, vone.v, (short)0, acc[4], false, false);
    acc[4] = __builtin_amdgcn_wmma_f32_16x16x32_f16(false, pa[1].v, false, vone.v, (short)0, acc[4], false, false);
  }

  // ---- normalize + store fp32 out [B,S,H,D] ------------------------------
#pragma unroll
  for (int r = 0; r < 8; ++r) {
    const float rinv = __builtin_amdgcn_rcpf(acc[4][r]);
    const int m = r + hi * 8;
    float* op = O + (((size_t)b * S_ + (m0 + m)) * H_ + h) * D_ + ln;
#pragma unroll
    for (int t = 0; t < 4; ++t)
      op[t * 16] = acc[t][r] * rinv;
  }
}

extern "C" void kernel_launch(void* const* d_in, const int* in_sizes, int n_in,
                              void* d_out, int out_size, void* d_ws, size_t ws_size,
                              hipStream_t stream) {
  (void)in_sizes; (void)n_in; (void)out_size; (void)d_ws; (void)ws_size;
  const float* Q = (const float*)d_in[0];
  const float* K = (const float*)d_in[1];
  const float* V = (const float*)d_in[2];
  float* O = (float*)d_out;
  dim3 grid(S_ / BLOCK_M, B_ * H_);
  dim3 block(256);
  hipLaunchKernelGGL(fa_fwd_kernel, grid, block, 0, stream, Q, K, V, O);
}
